// MLA_47201690583363
// MI455X (gfx1250) — matrix operand order repaired and checked by reading, not compiled, not run on hardware
//
#include <hip/hip_runtime.h>

// ---------------------------------------------------------------------------
// Problem constants (from reference)
// ---------------------------------------------------------------------------
#define BB    2
#define TT    2048
#define CDIM  2048
#define HH    16
#define QLRD  1024
#define KVLRD 512
#define DNN   128
#define DRR   64
#define DVV   128
#define DQKD  192          // DN + DR
#define MT    (BB * TT)    // 4096 tokens

// ---------------------------------------------------------------------------
// WMMA plumbing (gfx1250 / CDNA5, wave32)
// ---------------------------------------------------------------------------
typedef __attribute__((ext_vector_type(16))) __bf16        v16bf_t;
typedef __attribute__((ext_vector_type(8)))  float         v8f_t;
typedef __attribute__((ext_vector_type(4)))  unsigned int  u32x4_t;
typedef __attribute__((ext_vector_type(8)))  int           i32x8_t;
typedef __attribute__((ext_vector_type(4)))  int           i32x4_t;

union Frag {
    unsigned int u[8];
    v16bf_t      v;
};

__device__ __forceinline__ v8f_t wmma_bf16(const Frag& a, const Frag& b, v8f_t c) {
    // (neg_a, A, neg_b, B, c_mod, C, reuse_a, reuse_b)
    return __builtin_amdgcn_wmma_f32_16x16x32_bf16(false, a.v, false, b.v,
                                                   (short)0, c, false, false);
}

__device__ __forceinline__ unsigned short f2bf(float f) {
    unsigned int x = __builtin_bit_cast(unsigned int, f);
    unsigned int r = x + 0x7FFFu + ((x >> 16) & 1u);     // round-to-nearest-even
    if ((x & 0x7F800000u) == 0x7F800000u) r = x;         // keep inf/nan
    return (unsigned short)(r >> 16);
}

// ---------------------------------------------------------------------------
// Tensor Data Mover: 2D tile (global -> LDS), bf16 elements (data_size=2B).
// D# bitfields per CDNA5 ISA 8.3/8.4 (groups 2/3 zero: <=2D tensor).
// ---------------------------------------------------------------------------
__device__ __forceinline__ void tdm_load_2d(unsigned lds_off, const void* gptr,
                                            unsigned tile0, unsigned tile1,
                                            unsigned long long stride0,
                                            unsigned tdim0, unsigned tdim1)
{
    const unsigned long long ga = (unsigned long long)gptr;
    u32x4_t g0;
    g0.x = 1u;                                           // count=1, user D#
    g0.y = lds_off;                                      // lds_addr
    g0.z = (unsigned)(ga & 0xFFFFFFFFu);                 // global_addr[31:0]
    g0.w = (unsigned)((ga >> 32) & 0x01FFFFFFu)          // global_addr[56:32]
         | (2u << 30);                                   // type=2 ("image")
    i32x8_t g1;
    g1.s0 = (int)(1u << 16);                             // data_size=1 -> 2 bytes
    g1.s1 = (int)((tdim0 & 0xFFFFu) << 16);              // tensor_dim0[15:0]
    g1.s2 = (int)((tdim0 >> 16) | ((tdim1 & 0xFFFFu) << 16));
    g1.s3 = (int)((tdim1 >> 16) | ((tile0 & 0xFFFFu) << 16));   // tile_dim0
    g1.s4 = (int)(tile1 & 0xFFFFu);                      // tile_dim1 (tile_dim2=0)
    g1.s5 = (int)(stride0 & 0xFFFFFFFFu);                // dim0 stride
    g1.s6 = (int)((stride0 >> 32) & 0xFFFFu);            // stride hi (stride1=0)
    g1.s7 = 0;
    const i32x4_t z4 = {0, 0, 0, 0};
#if __clang_major__ >= 23
    const i32x8_t z8 = {0, 0, 0, 0, 0, 0, 0, 0};
    __builtin_amdgcn_tensor_load_to_lds(g0, g1, z4, z4, z8, 0);
#else
    __builtin_amdgcn_tensor_load_to_lds(g0, g1, z4, z4, 0);
#endif
}

// ---------------------------------------------------------------------------
// fp32 -> bf16 elementwise conversion
// ---------------------------------------------------------------------------
__global__ __launch_bounds__(256)
void cvt_bf16_kernel(const float* __restrict__ src, unsigned short* __restrict__ dst, long n) {
    long i = (long)blockIdx.x * blockDim.x + threadIdx.x;
    long stride = (long)gridDim.x * blockDim.x;
    for (; i < n; i += stride) dst[i] = f2bf(src[i]);
}

// strided slice (first dcols of each scols-row) -> bf16
__global__ __launch_bounds__(256)
void slice_cvt_bf16_kernel(const float* __restrict__ src, unsigned short* __restrict__ dst,
                           long rows, long scols, long dcols) {
    long n = rows * dcols;
    long stride = (long)gridDim.x * blockDim.x;
    for (long i = (long)blockIdx.x * blockDim.x + threadIdx.x; i < n; i += stride) {
        long rr = i / dcols, cc = i - rr * dcols;
        dst[i] = f2bf(src[rr * scols + cc]);
    }
}

// ---------------------------------------------------------------------------
// GEMM: C[M,N] = A[M,K](bf16) * W[N,K](bf16)^T + bias[N]   (einsum '...i,oi->...o')
// 8 waves/block stacked along M; wave tile 32x64 (2 A-frags x 4 B-cols = 8 WMMA
// per 32-wide k-step). k-loop unrolled by 2 with ping-pong X/Y fragment buffers
// so the software pipeline needs no register copies.
// Block tile 256x64; grid (N/64, M/256). Requires K % 64 == 0.
// ---------------------------------------------------------------------------
__global__ __launch_bounds__(256)
void gemm_bf16_wmma(const unsigned short* __restrict__ A,
                    const unsigned short* __restrict__ W,
                    const float* __restrict__ bias,
                    float* __restrict__ C,
                    int M, int N, int K)
{
    const int lane = threadIdx.x & 31;
    const int wave = threadIdx.x >> 5;
    const int row0 = blockIdx.y * 256 + wave * 32;
    const int col0 = blockIdx.x * 64;
    const int r    = lane & 15;
    const int hi   = (lane >> 4) & 1;
    const int akb  = hi ? 16 : 0;   // A-fragment K base per lane half
    const int bkb  = hi ? 16 : 0;   // B-fragment K base per lane half

    const unsigned short* Ar0 = A + (size_t)(row0 + r) * K + akb;
    const unsigned short* Ar1 = A + (size_t)(row0 + 16 + r) * K + akb;
    const unsigned short* Wr[4];
#pragma unroll
    for (int t = 0; t < 4; ++t)
        Wr[t] = W + (size_t)(col0 + t * 16 + r) * K + bkb;

    v8f_t acc[2][4] = {};
    Frag aX[2], bX[4], aY[2], bY[4];

    auto loadA = [&](Frag& f, const unsigned short* p) {
#pragma unroll
        for (int v = 0; v < 4; ++v) {
            f.u[v]     = *reinterpret_cast<const unsigned int*>(p + 2 * v);
            f.u[v + 4] = *reinterpret_cast<const unsigned int*>(p + 8 + 2 * v);
        }
    };
    auto loadB = [&](Frag& f, const unsigned short* p) {
#pragma unroll
        for (int v = 0; v < 8; ++v)
            f.u[v] = *reinterpret_cast<const unsigned int*>(p + 2 * v);
    };

    // prologue: fill X with k=0
    loadA(aX[0], Ar0);
    loadA(aX[1], Ar1);
#pragma unroll
    for (int t = 0; t < 4; ++t) loadB(bX[t], Wr[t]);

    for (int k = 0; k < K; k += 64) {
        // issue Y loads (k+32), then consume X (k)
        loadA(aY[0], Ar0 + k + 32);
        loadA(aY[1], Ar1 + k + 32);
#pragma unroll
        for (int t = 0; t < 4; ++t) loadB(bY[t], Wr[t] + k + 32);
        __builtin_prefetch(Ar0 + k + 64, 0, 3);
        __builtin_prefetch(Ar1 + k + 64, 0, 3);
#pragma unroll
        for (int t = 0; t < 4; ++t) {
            acc[0][t] = wmma_bf16(aX[0], bX[t], acc[0][t]);
            acc[1][t] = wmma_bf16(aX[1], bX[t], acc[1][t]);
        }
        // issue X loads (k+64), then consume Y (k+32)
        if (k + 64 < K) {
            loadA(aX[0], Ar0 + k + 64);
            loadA(aX[1], Ar1 + k + 64);
#pragma unroll
            for (int t = 0; t < 4; ++t) loadB(bX[t], Wr[t] + k + 64);
        }
#pragma unroll
        for (int t = 0; t < 4; ++t) {
            acc[0][t] = wmma_bf16(aY[0], bY[t], acc[0][t]);
            acc[1][t] = wmma_bf16(aY[1], bY[t], acc[1][t]);
        }
    }

#pragma unroll
    for (int g = 0; g < 2; ++g) {
#pragma unroll
        for (int t = 0; t < 4; ++t) {
            const int cc = col0 + t * 16 + r;
            const float bv = bias[cc];
#pragma unroll
            for (int i = 0; i < 8; ++i) {
                const int rr = row0 + g * 16 + i + hi * 8;   // C-layout row striping
                C[(size_t)rr * N + cc] = acc[g][t][i] + bv;
            }
        }
    }
}

// ---------------------------------------------------------------------------
// Row LayerNorm (fp32 stats) -> bf16
// ---------------------------------------------------------------------------
__global__ __launch_bounds__(256)
void layernorm_bf16_kernel(const float* __restrict__ X, const float* __restrict__ g,
                           const float* __restrict__ beta, unsigned short* __restrict__ Y,
                           int N)
{
    __shared__ float s1[256], s2[256];
    const int row = blockIdx.x;
    const float* x = X + (size_t)row * N;
    float sum = 0.f, sq = 0.f;
    for (int i = threadIdx.x; i < N; i += 256) { float v = x[i]; sum += v; sq += v * v; }
    s1[threadIdx.x] = sum; s2[threadIdx.x] = sq;
    __syncthreads();
    for (int s = 128; s > 0; s >>= 1) {
        if ((int)threadIdx.x < s) {
            s1[threadIdx.x] += s1[threadIdx.x + s];
            s2[threadIdx.x] += s2[threadIdx.x + s];
        }
        __syncthreads();
    }
    const float mu   = s1[0] / (float)N;
    const float var  = s2[0] / (float)N - mu * mu;
    const float rstd = rsqrtf(var + 1e-5f);
    unsigned short* y = Y + (size_t)row * N;
    for (int i = threadIdx.x; i < N; i += 256)
        y[i] = f2bf((x[i] - mu) * rstd * g[i] + beta[i]);
}

// ---------------------------------------------------------------------------
// Build query/key (with RoPE) and transposed V.
//   Qb,Kb : [B,H,T,192] bf16 ; Vt : [B,H,128,T] bf16 (key dim contiguous)
// grid (T, H, B), 192 threads.
// ---------------------------------------------------------------------------
__global__ __launch_bounds__(192)
void build_qkv_kernel(const float* __restrict__ qfull,  // [B*T, H*192]
                      const float* __restrict__ kvf,    // [B*T, 576]
                      const float* __restrict__ kvb,    // [B*T, H*256]
                      unsigned short* __restrict__ Qb,
                      unsigned short* __restrict__ Kb,
                      unsigned short* __restrict__ Vt)
{
    const int t = blockIdx.x, h = blockIdx.y, b = blockIdx.z, d = threadIdx.x;
    const size_t bt  = (size_t)b * TT + t;
    const size_t bh  = (size_t)b * HH + h;
    const size_t bht = bh * TT + t;
    const float* qrow = qfull + bt * (size_t)(HH * DQKD) + (size_t)h * DQKD;
    const float* krow = kvb   + bt * (size_t)(HH * (DNN + DVV)) + (size_t)h * (DNN + DVV);

    float qv, kv;
    if (d < DNN) {
        qv = qrow[DRR + d];   // q_nope -> query[0:128]
        kv = krow[d];         // k_nope -> key[0:128]
    } else {
        const int j = d - DNN;
        const int i = (j < 32) ? j : (j - 32);
        const float invf = __powf(10000.f, -(2.0f * (float)i) / 64.0f);
        const float ang  = (float)t * invf;
        const float c = __cosf(ang), s = __sinf(ang);
        if (j < 32) qv = qrow[i] * c - qrow[i + 32] * s;
        else        qv = qrow[i + 32] * c + qrow[i] * s;
        const float* kr = kvf + bt * 576 + KVLRD;  // shared rope key, same per head
        if (j < 32) kv = kr[i] * c - kr[i + 32] * s;
        else        kv = kr[i + 32] * c + kr[i] * s;
    }
    Qb[bht * DQKD + d] = f2bf(qv);
    Kb[bht * DQKD + d] = f2bf(kv);
    if (d < DVV)
        Vt[(bh * DVV + d) * TT + t] = f2bf(krow[DNN + d]);
}

// ---------------------------------------------------------------------------
// Flash attention, TDM-staged:
//   All 8 waves share 32-key chunks. Wave 0 DMAs the K tile (32x192) and the
//   V^T tile (128x32) into LDS via tensor_load_to_lds, s_wait_tensorcnt, then
//   every wave does S = Q K^T (12 wmma, B-frags from LDS), online softmax,
//   P re-layout through LDS, and PV (8 wmma, V-frags from LDS).
// grid (T/128, B*H), 256 threads (8 waves).
// ---------------------------------------------------------------------------
__global__ __launch_bounds__(256)
void mla_attention_kernel(const unsigned short* __restrict__ Q,
                          const unsigned short* __restrict__ Kc,
                          const unsigned short* __restrict__ Vt,
                          float* __restrict__ Y)          // [B, T, H*128]
{
    __shared__ unsigned short kstage[32 * DQKD];    // 12 KB : keys x d
    __shared__ unsigned short vstage[DVV * 32];     //  8 KB : d x keys
    __shared__ unsigned short pstage[8][16 * 32];   //  8 KB : per-wave P re-layout

    const int lane  = threadIdx.x & 31;
    const int wave  = threadIdx.x >> 5;
    const int q0blk = blockIdx.x * 128;
    const int q0    = q0blk + wave * 16;
    const int bh    = blockIdx.y;
    const int b     = bh >> 4, h = bh & 15;
    const int r     = lane & 15;
    const int hi    = (lane >> 4) & 1;
    const int akb   = hi ? 16 : 0;
    const int bkb   = hi ? 16 : 0;

    const unsigned short* Qb = Q  + (size_t)bh * TT * DQKD;
    const unsigned short* Kb = Kc + (size_t)bh * TT * DQKD;
    const unsigned short* Vb = Vt + (size_t)bh * DVV * TT;

    const unsigned kst_off = (unsigned)(unsigned long long)(const void*)kstage;
    const unsigned vst_off = (unsigned)(unsigned long long)(const void*)vstage;

    // Preload the 16x192 Q tile as six A-fragments.
    Frag qf[6];
    const unsigned short* qrow = Qb + (size_t)(q0 + r) * DQKD;
#pragma unroll
    for (int dc = 0; dc < 6; ++dc) {
#pragma unroll
        for (int v = 0; v < 4; ++v) {
            qf[dc].u[v]     = *reinterpret_cast<const unsigned int*>(qrow + dc * 32 + akb + 2 * v);
            qf[dc].u[v + 4] = *reinterpret_cast<const unsigned int*>(qrow + dc * 32 + akb + 8 + 2 * v);
        }
    }

    v8f_t O[8] = {};
    float mrow[8], lrow[8];
#pragma unroll
    for (int i = 0; i < 8; ++i) { mrow[i] = -1e30f; lrow[i] = 0.f; }

    const float scale  = 0.07216878364f;             // 1/sqrt(192)
    const int   nchunk = (q0blk + 128) / 32;         // block-shared causal bound

    for (int c = 0; c < nchunk; ++c) {
        const int kbase = c * 32;
        __syncthreads();                             // previous stage fully consumed
        if (wave == 0) {
            // K tile: rows kbase..kbase+31, 192 elems each, row stride 192
            tdm_load_2d(kst_off, Kb + (size_t)kbase * DQKD,
                        DQKD, 32, DQKD, DQKD, (unsigned)(TT - kbase));
            // V^T tile: 128 d-rows, 32 keys each, row stride TT
            tdm_load_2d(vst_off, Vb + kbase,
                        32, DVV, TT, (unsigned)(TT - kbase), DVV);
            __builtin_amdgcn_s_wait_tensorcnt(0);
        }
        __syncthreads();                             // stage visible to all waves
        if (kbase > q0 + 15) continue;               // fully masked for this wave

        v8f_t s0 = {}, s1 = {};
#pragma unroll
        for (int dc = 0; dc < 6; ++dc) {             // keys kbase..kbase+15
            Frag bf;
            const unsigned short* kr = kstage + (size_t)r * DQKD + dc * 32 + bkb;
#pragma unroll
            for (int v = 0; v < 8; ++v)
                bf.u[v] = *reinterpret_cast<const unsigned int*>(kr + 2 * v);
            s0 = wmma_bf16(qf[dc], bf, s0);
        }
#pragma unroll
        for (int dc = 0; dc < 6; ++dc) {             // keys kbase+16..kbase+31
            Frag bf;
            const unsigned short* kr = kstage + (size_t)(16 + r) * DQKD + dc * 32 + bkb;
#pragma unroll
            for (int v = 0; v < 8; ++v)
                bf.u[v] = *reinterpret_cast<const unsigned int*>(kr + 2 * v);
            s1 = wmma_bf16(qf[dc], bf, s1);
        }

        // causal mask + online softmax (row i of VGPR slot maps to rows i / i+8)
        float p0[8], p1[8];
#pragma unroll
        for (int i = 0; i < 8; ++i) {
            const int q = q0 + i + hi * 8;
            float v0 = s0[i] * scale;
            float v1 = s1[i] * scale;
            if (kbase + r > q)      v0 = -1e30f;
            if (kbase + 16 + r > q) v1 = -1e30f;
            float tmax = fmaxf(v0, v1);
#pragma unroll
            for (int m = 1; m < 16; m <<= 1) tmax = fmaxf(tmax, __shfl_xor(tmax, m, 32));
            const float mnew  = fmaxf(mrow[i], tmax);
            const float alpha = __expf(mrow[i] - mnew);
            mrow[i] = mnew;
            const float e0 = __expf(v0 - mnew);
            const float e1 = __expf(v1 - mnew);
            float tsum = e0 + e1;
#pragma unroll
            for (int m = 1; m < 16; m <<= 1) tsum += __shfl_xor(tsum, m, 32);
            lrow[i] = lrow[i] * alpha + tsum;
            p0[i] = e0; p1[i] = e1;
#pragma unroll
            for (int nt = 0; nt < 8; ++nt) O[nt][i] *= alpha;
        }

        // Re-layout P: C-format -> A-format via per-wave LDS staging.
#pragma unroll
        for (int i = 0; i < 8; ++i) {
            const int row = i + hi * 8;
            pstage[wave][row * 32 + r]      = f2bf(p0[i]);
            pstage[wave][row * 32 + 16 + r] = f2bf(p1[i]);
        }
        asm volatile("s_wait_dscnt 0x0" ::: "memory");
        Frag pf;
#pragma unroll
        for (int v = 0; v < 4; ++v) {
            pf.u[v]     = *reinterpret_cast<const unsigned int*>(&pstage[wave][r * 32 + akb + 2 * v]);
            pf.u[v + 4] = *reinterpret_cast<const unsigned int*>(&pstage[wave][r * 32 + akb + 8 + 2 * v]);
        }

        // O += P * V   (V-fragments from LDS, key dim contiguous)
#pragma unroll
        for (int nt = 0; nt < 8; ++nt) {
            Frag vf;
            const unsigned short* vr = vstage + (size_t)(nt * 16 + r) * 32 + bkb;
#pragma unroll
            for (int v = 0; v < 8; ++v)
                vf.u[v] = *reinterpret_cast<const unsigned int*>(vr + 2 * v);
            O[nt] = wmma_bf16(pf, vf, O[nt]);
        }
    }

    // Epilogue: divide by row sums, write [b, t, h*128 + d]
#pragma unroll
    for (int i = 0; i < 8; ++i) {
        const float inv = 1.0f / lrow[i];
        const int q = q0 + i + hi * 8;
        float* yrow = Y + ((size_t)b * TT + q) * (size_t)(HH * DVV) + (size_t)h * DVV;
#pragma unroll
        for (int nt = 0; nt < 8; ++nt)
            yrow[nt * 16 + r] = O[nt][i] * inv;
    }
}

// ---------------------------------------------------------------------------
// Host launcher
// ---------------------------------------------------------------------------
extern "C" void kernel_launch(void* const* d_in, const int* in_sizes, int n_in,
                              void* d_out, int out_size, void* d_ws, size_t ws_size,
                              hipStream_t stream)
{
    (void)in_sizes; (void)n_in; (void)out_size; (void)ws_size;

    const float* x     = (const float*)d_in[0];
    const float* wq_a  = (const float*)d_in[1];
    const float* bq_a  = (const float*)d_in[2];
    const float* g_q   = (const float*)d_in[3];
    const float* b_q   = (const float*)d_in[4];
    const float* wq_b  = (const float*)d_in[5];
    const float* bq_b  = (const float*)d_in[6];
    const float* wkv_a = (const float*)d_in[7];
    const float* bkv_a = (const float*)d_in[8];
    const float* wkv_b = (const float*)d_in[9];
    const float* bkv_b = (const float*)d_in[10];
    const float* wout  = (const float*)d_in[11];
    const float* bout  = (const float*)d_in[12];

    char* ws = (char*)d_ws;
    size_t off = 0;
    auto carve = [&](size_t bytes) -> char* {
        char* p = ws + off;
        off = (off + bytes + 255) & ~(size_t)255;
        return p;
    };

    unsigned short* x_bf     = (unsigned short*)carve((size_t)MT * CDIM * 2);
    unsigned short* wqa_bf   = (unsigned short*)carve((size_t)QLRD * CDIM * 2);
    unsigned short* wqb_bf   = (unsigned short*)carve((size_t)HH * DQKD * QLRD * 2);
    unsigned short* wkva_bf  = (unsigned short*)carve((size_t)(KVLRD + DRR) * CDIM * 2);
    unsigned short* wkvb_bf  = (unsigned short*)carve((size_t)HH * (DNN + DVV) * KVLRD * 2);
    unsigned short* wout_bf  = (unsigned short*)carve((size_t)CDIM * CDIM * 2);
    float*          q_pre    = (float*)carve((size_t)MT * QLRD * 4);
    unsigned short* qlora_bf = (unsigned short*)carve((size_t)MT * QLRD * 2);
    float*          qfull    = (float*)carve((size_t)MT * HH * DQKD * 4);
    float*          kvf      = (float*)carve((size_t)MT * (KVLRD + DRR) * 4);
    unsigned short* kvlor_bf = (unsigned short*)carve((size_t)MT * KVLRD * 2);
    float*          kvbf     = (float*)carve((size_t)MT * HH * (DNN + DVV) * 4);
    unsigned short* Qb       = (unsigned short*)carve((size_t)BB * HH * TT * DQKD * 2);
    unsigned short* Kb       = (unsigned short*)carve((size_t)BB * HH * TT * DQKD * 2);
    unsigned short* Vt       = (unsigned short*)carve((size_t)BB * HH * DVV * TT * 2);
    float*          yf       = (float*)carve((size_t)MT * CDIM * 4);
    unsigned short* y_bf     = (unsigned short*)carve((size_t)MT * CDIM * 2);

    auto cvt = [&](const float* s, unsigned short* d, long n) {
        long blocks = (n + 1023) / 1024;
        if (blocks > 4096) blocks = 4096;
        cvt_bf16_kernel<<<dim3((unsigned)blocks), dim3(256), 0, stream>>>(s, d, n);
    };

    // 0) bf16 copies of x and weights
    cvt(x,     x_bf,    (long)MT * CDIM);
    cvt(wq_a,  wqa_bf,  (long)QLRD * CDIM);
    cvt(wq_b,  wqb_bf,  (long)HH * DQKD * QLRD);
    cvt(wkv_a, wkva_bf, (long)(KVLRD + DRR) * CDIM);
    cvt(wkv_b, wkvb_bf, (long)HH * (DNN + DVV) * KVLRD);
    cvt(wout,  wout_bf, (long)CDIM * CDIM);

    // 1) q_pre = x @ wq_a^T + bq_a           [4096 x 1024]
    gemm_bf16_wmma<<<dim3(QLRD / 64, MT / 256), dim3(256), 0, stream>>>(
        x_bf, wqa_bf, bq_a, q_pre, MT, QLRD, CDIM);

    // 2) q_lora = LN(q_pre) * g + b  -> bf16
    layernorm_bf16_kernel<<<dim3(MT), dim3(256), 0, stream>>>(q_pre, g_q, b_q, qlora_bf, QLRD);

    // 3) q = q_lora @ wq_b^T + bq_b          [4096 x 3072]
    gemm_bf16_wmma<<<dim3((HH * DQKD) / 64, MT / 256), dim3(256), 0, stream>>>(
        qlora_bf, wqb_bf, bq_b, qfull, MT, HH * DQKD, QLRD);

    // 4) kv = x @ wkv_a^T + bkv_a            [4096 x 576]
    gemm_bf16_wmma<<<dim3((KVLRD + DRR) / 64, MT / 256), dim3(256), 0, stream>>>(
        x_bf, wkva_bf, bkv_a, kvf, MT, KVLRD + DRR, CDIM);

    // 5) kv_lora (first 512 cols) -> bf16
    slice_cvt_bf16_kernel<<<dim3(2048), dim3(256), 0, stream>>>(
        kvf, kvlor_bf, (long)MT, (long)(KVLRD + DRR), (long)KVLRD);

    // 6) kvb = kv_lora @ wkv_b^T + bkv_b     [4096 x 4096]
    gemm_bf16_wmma<<<dim3((HH * (DNN + DVV)) / 64, MT / 256), dim3(256), 0, stream>>>(
        kvlor_bf, wkvb_bf, bkv_b, kvbf, MT, HH * (DNN + DVV), KVLRD);

    // 7) RoPE + split/concat -> Q, K, V^T (bf16)
    build_qkv_kernel<<<dim3(TT, HH, BB), dim3(192), 0, stream>>>(
        qfull, kvf, kvbf, Qb, Kb, Vt);

    // 8) causal flash attention (TDM-staged) -> y fp32 [b, t, h*128]
    mla_attention_kernel<<<dim3(TT / 128, BB * HH), dim3(256), 0, stream>>>(Qb, Kb, Vt, yf);

    // 9) y -> bf16, out = y @ wout^T + bout  [4096 x 2048] fp32 into d_out
    cvt(yf, y_bf, (long)MT * CDIM);
    gemm_bf16_wmma<<<dim3(CDIM / 64, MT / 256), dim3(256), 0, stream>>>(
        y_bf, wout_bf, bout, (float*)d_out, MT, CDIM, CDIM);
}
